// HyperElasticityModel_30897994728214
// MI455X (gfx1250) — matrix operand order repaired and checked by reading, not compiled, not run on hardware
//
#include <hip/hip_runtime.h>

typedef __attribute__((ext_vector_type(16))) _Float16 v16h;
typedef __attribute__((ext_vector_type(2)))  _Float16 h2;
typedef __attribute__((ext_vector_type(8)))  float    v8f;

static constexpr int Nn  = 16384;
static constexpr int EMn = 98304;
static constexpr int EWn = 8192;
static constexpr int Sn  = 15;

// ---------------------------------------------------------------------------
// Weight prep: f32 row-major [K x N] -> WMMA B-fragment f16 layout.
// Fragment (kc,nc) = 32x16 tile; lane l (half=l>>4, nl=l&15) holds 16 halves:
//   elem i = B[kc*32 + (i < 8 ? half*8 + i : 16 + half*8 + (i-8))][nc*16 + nl]
// Stored contiguously: frag[((kc*NC+nc)*32 + lane)*16 + i]
// ---------------------------------------------------------------------------
__global__ void prep_w(const float* W, _Float16* frag, int Kact, int Nact,
                       int KC, int NC, size_t wstride, size_t fstride, size_t total) {
  size_t tid = (size_t)blockIdx.x * blockDim.x + threadIdx.x;
  if (tid >= total) return;
  size_t per = (size_t)KC * NC * 512;
  int s = (int)(tid / per);
  size_t r = tid % per;
  int i    = (int)(r & 15);
  int lane = (int)((r >> 4) & 31);
  int nc   = (int)((r >> 9) % NC);
  int kc   = (int)(r / (512 * (size_t)NC));
  int half = lane >> 4, nl = lane & 15;
  int k = kc * 32 + ((i < 8) ? (half * 8 + i) : (16 + half * 8 + (i - 8)));
  int n = nc * 16 + nl;
  float v = 0.f;
  if (k < Kact && n < Nact) v = W[(size_t)s * wstride + (size_t)k * Nact + n];
  frag[(size_t)s * fstride + (((size_t)kc * NC + nc) * 32 + lane) * 16 + i] = (_Float16)v;
}

// ---------------------------------------------------------------------------
// Feature builders (normalized, padded to 32 f16 columns)
// ---------------------------------------------------------------------------
__global__ void node_feat(const float* wp, const float* twp, const int* nt,
                          const float* mean, const float* stdv, _Float16* out) {
  int i = blockIdx.x * blockDim.x + threadIdx.x;
  if (i >= Nn) return;
  int t = nt[i];
  float f[12];
  float m = (t == 0) ? 1.f : 0.f;
  f[0] = m * (twp[i*3+0] - wp[i*3+0]);
  f[1] = m * (twp[i*3+1] - wp[i*3+1]);
  f[2] = m * (twp[i*3+2] - wp[i*3+2]);
  #pragma unroll
  for (int j = 0; j < 9; ++j) f[3+j] = (t == j) ? 1.f : 0.f;
  _Float16* o = out + (size_t)i * 32;
  #pragma unroll
  for (int c = 0; c < 12; ++c) o[c] = (_Float16)((f[c] - mean[c]) / stdv[c]);
  #pragma unroll
  for (int c = 12; c < 32; ++c) o[c] = (_Float16)0.f;
}

__global__ void mesh_feat(const float* wp, const float* mp, const int* srcs, const int* dsts,
                          const float* mean, const float* stdv, _Float16* out) {
  int e = blockIdx.x * blockDim.x + threadIdx.x;
  if (e >= EMn) return;
  int s = srcs[e], d = dsts[e];
  float f[8];
  float a0 = wp[s*3+0]-wp[d*3+0], a1 = wp[s*3+1]-wp[d*3+1], a2 = wp[s*3+2]-wp[d*3+2];
  float b0 = mp[s*3+0]-mp[d*3+0], b1 = mp[s*3+1]-mp[d*3+1], b2 = mp[s*3+2]-mp[d*3+2];
  f[0]=a0; f[1]=a1; f[2]=a2; f[3]=sqrtf(a0*a0+a1*a1+a2*a2);
  f[4]=b0; f[5]=b1; f[6]=b2; f[7]=sqrtf(b0*b0+b1*b1+b2*b2);
  _Float16* o = out + (size_t)e * 32;
  #pragma unroll
  for (int c = 0; c < 8; ++c) o[c] = (_Float16)((f[c] - mean[c]) / stdv[c]);
  #pragma unroll
  for (int c = 8; c < 32; ++c) o[c] = (_Float16)0.f;
}

__global__ void world_feat(const float* wp, const int* srcs, const int* dsts,
                           const float* mean, const float* stdv, _Float16* out) {
  int e = blockIdx.x * blockDim.x + threadIdx.x;
  if (e >= EWn) return;
  int s = srcs[e], d = dsts[e];
  float a0 = wp[s*3+0]-wp[d*3+0], a1 = wp[s*3+1]-wp[d*3+1], a2 = wp[s*3+2]-wp[d*3+2];
  float f[4] = {a0, a1, a2, sqrtf(a0*a0+a1*a1+a2*a2)};
  _Float16* o = out + (size_t)e * 32;
  #pragma unroll
  for (int c = 0; c < 4; ++c) o[c] = (_Float16)((f[c] - mean[c]) / stdv[c]);
  #pragma unroll
  for (int c = 4; c < 32; ++c) o[c] = (_Float16)0.f;
}

__global__ void zero_f32(float* p, size_t n) {
  size_t i = (size_t)blockIdx.x * blockDim.x + threadIdx.x;
  size_t st = (size_t)gridDim.x * blockDim.x;
  for (; i < n; i += st) p[i] = 0.f;
}

__global__ void cvt_f32_h(const float* in, _Float16* o, size_t n) {
  size_t i = (size_t)blockIdx.x * blockDim.x + threadIdx.x;
  size_t st = (size_t)gridDim.x * blockDim.x;
  for (; i < n; i += st) o[i] = (_Float16)in[i];
}

// ---------------------------------------------------------------------------
// Fused 3-layer MLP + LayerNorm (+residual, +scatter-add aggregation).
// One wave = 16 rows x 128 cols (8 WMMA accumulators). 8 waves/block.
// Layer0 K = KC0*32 (segments of 128 cols with gather for SEG3 mode).
// ---------------------------------------------------------------------------
template<int KC0, bool SEG3, bool RESID, bool AGG>
__global__ __launch_bounds__(256) void mlp_fused(
    const _Float16* s0, int ld0, const _Float16* s1, const _Float16* s2,
    const int* g1, const int* g2,
    const _Float16* w0f, const _Float16* w1f, const _Float16* w2f,
    const float* b0, const float* b1, const float* b2,
    const float* gam, const float* bet,
    float* st_f32, _Float16* st_h16,
    float* agg, const int* aggidx) {
  __shared__ _Float16 sH[8][16 * 128];
  const int lane = threadIdx.x & 31;
  const int wave = threadIdx.x >> 5;
  const int nl   = lane & 15;
  const int half = lane >> 4;
  const int tilerow = blockIdx.x * 128 + wave * 16;
  const int arow = tilerow + nl;           // A-operand row this lane supplies
  _Float16* hp = sH[wave];

  const _Float16* rb0 = s0 + (size_t)arow * ld0;
  const _Float16* rb1 = nullptr;
  const _Float16* rb2 = nullptr;
  if (SEG3) {
    int r1 = g1 ? g1[arow] : arow;
    int r2 = g2 ? g2[arow] : arow;
    rb1 = s1 + (size_t)r1 * 128;
    rb2 = s2 + (size_t)r2 * 128;
  }

  const v8f vz = {};
  v8f acc[8];
  #pragma unroll
  for (int nc = 0; nc < 8; ++nc) acc[nc] = vz;

  // -------- layer 0: X @ W0  (A from global, possibly gathered) --------
  #pragma unroll
  for (int kc = 0; kc < KC0; ++kc) {
    const _Float16* base;
    if (SEG3) {
      const _Float16* rb = (kc < 4) ? rb0 : ((kc < 8) ? rb1 : rb2);
      base = rb + (kc & 3) * 32;
    } else {
      base = rb0 + kc * 32;
    }
    const h2* ap = (const h2*)base;
    v16h a;
    #pragma unroll
    for (int j = 0; j < 8; ++j) {   // 16-bit A layout: K pairs per dword
      h2 t = ap[j + 4 * half + ((j >= 4) ? 4 : 0)];
      a[2*j] = t.x; a[2*j+1] = t.y;
    }
    const v16h* bp = (const v16h*)w0f + ((size_t)kc * 8) * 32 + lane;
    #pragma unroll
    for (int nc = 0; nc < 8; ++nc) {
      v16h b = bp[(size_t)nc * 32];
      acc[nc] = __builtin_amdgcn_wmma_f32_16x16x32_f16(false, a, false, b,
                                                       (short)0, acc[nc], false, false);
    }
  }
  // bias + relu -> LDS h-tile (f16, row-major [16][128])
  #pragma unroll
  for (int nc = 0; nc < 8; ++nc) {
    int n = nc * 16 + nl;
    float bb = b0[n];
    #pragma unroll
    for (int r = 0; r < 8; ++r) {
      float v = acc[nc][r] + bb;
      v = v > 0.f ? v : 0.f;
      hp[(r + 8 * half) * 128 + n] = (_Float16)v;
    }
  }

  // -------- layer 1: h @ W1 (A from LDS) --------
  #pragma unroll
  for (int nc = 0; nc < 8; ++nc) acc[nc] = vz;
  #pragma unroll
  for (int kc = 0; kc < 4; ++kc) {
    v16h a;
    #pragma unroll
    for (int j = 0; j < 8; ++j) {
      int k = kc * 32 + ((j < 4) ? (2*j + 8*half) : (16 + 2*(j-4) + 8*half));
      a[2*j]   = hp[nl * 128 + k];
      a[2*j+1] = hp[nl * 128 + k + 1];
    }
    const v16h* bp = (const v16h*)w1f + ((size_t)kc * 8) * 32 + lane;
    #pragma unroll
    for (int nc = 0; nc < 8; ++nc) {
      v16h b = bp[(size_t)nc * 32];
      acc[nc] = __builtin_amdgcn_wmma_f32_16x16x32_f16(false, a, false, b,
                                                       (short)0, acc[nc], false, false);
    }
  }
  #pragma unroll
  for (int nc = 0; nc < 8; ++nc) {
    int n = nc * 16 + nl;
    float bb = b1[n];
    #pragma unroll
    for (int r = 0; r < 8; ++r) {
      float v = acc[nc][r] + bb;
      v = v > 0.f ? v : 0.f;
      hp[(r + 8 * half) * 128 + n] = (_Float16)v;
    }
  }

  // -------- layer 2: h @ W2 + b2 (linear) --------
  #pragma unroll
  for (int nc = 0; nc < 8; ++nc) acc[nc] = vz;
  #pragma unroll
  for (int kc = 0; kc < 4; ++kc) {
    v16h a;
    #pragma unroll
    for (int j = 0; j < 8; ++j) {
      int k = kc * 32 + ((j < 4) ? (2*j + 8*half) : (16 + 2*(j-4) + 8*half));
      a[2*j]   = hp[nl * 128 + k];
      a[2*j+1] = hp[nl * 128 + k + 1];
    }
    const v16h* bp = (const v16h*)w2f + ((size_t)kc * 8) * 32 + lane;
    #pragma unroll
    for (int nc = 0; nc < 8; ++nc) {
      v16h b = bp[(size_t)nc * 32];
      acc[nc] = __builtin_amdgcn_wmma_f32_16x16x32_f16(false, a, false, b,
                                                       (short)0, acc[nc], false, false);
    }
  }
  #pragma unroll
  for (int nc = 0; nc < 8; ++nc) {
    float bb = b2[nc * 16 + nl];
    #pragma unroll
    for (int r = 0; r < 8; ++r) acc[nc][r] += bb;
  }

  // -------- LayerNorm stats via cross-lane reduction (per row) --------
  float mu[8], rs[8];
  #pragma unroll
  for (int r = 0; r < 8; ++r) {
    float s = 0.f;
    #pragma unroll
    for (int nc = 0; nc < 8; ++nc) s += acc[nc][r];
    s += __shfl_xor(s, 1); s += __shfl_xor(s, 2);
    s += __shfl_xor(s, 4); s += __shfl_xor(s, 8);
    mu[r] = s * (1.f / 128.f);
  }
  #pragma unroll
  for (int r = 0; r < 8; ++r) {
    float s = 0.f;
    #pragma unroll
    for (int nc = 0; nc < 8; ++nc) { float d = acc[nc][r] - mu[r]; s += d * d; }
    s += __shfl_xor(s, 1); s += __shfl_xor(s, 2);
    s += __shfl_xor(s, 4); s += __shfl_xor(s, 8);
    rs[r] = rsqrtf(s * (1.f / 128.f) + 1e-5f);
  }

  // -------- epilogue: LN * g + be, +residual, dual store, scatter-add ------
  float gl[8], bl[8];
  #pragma unroll
  for (int nc = 0; nc < 8; ++nc) { int n = nc*16 + nl; gl[nc] = gam[n]; bl[nc] = bet[n]; }
  #pragma unroll
  for (int r = 0; r < 8; ++r) {
    int grow = tilerow + r + 8 * half;
    size_t rbase = (size_t)grow * 128;
    size_t abase = 0;
    if (AGG) abase = (size_t)aggidx[grow] * 128;
    #pragma unroll
    for (int nc = 0; nc < 8; ++nc) {
      int n = nc * 16 + nl;
      float v = (acc[nc][r] - mu[r]) * rs[r] * gl[nc] + bl[nc];
      float res = RESID ? (st_f32[rbase + n] + v) : v;
      st_f32[rbase + n] = res;
      st_h16[rbase + n] = (_Float16)res;
      if (AGG) atomicAdd(&agg[abase + n], res);
    }
  }
}

// ---------------------------------------------------------------------------
// Decoder: relu(128->128) -> relu(128->128) -> 128->3 (N padded to 16), unnorm
// ---------------------------------------------------------------------------
__global__ __launch_bounds__(256) void dec_kernel(
    const _Float16* s0,
    const _Float16* w0f, const _Float16* w1f, const _Float16* w2f,
    const float* b0, const float* b1, const float* b2,
    const float* omean, const float* ostd, float* out) {
  __shared__ _Float16 sH[8][16 * 128];
  const int lane = threadIdx.x & 31;
  const int wave = threadIdx.x >> 5;
  const int nl   = lane & 15;
  const int half = lane >> 4;
  const int tilerow = blockIdx.x * 128 + wave * 16;
  const int arow = tilerow + nl;
  _Float16* hp = sH[wave];
  const _Float16* rb0 = s0 + (size_t)arow * 128;

  const v8f vz = {};
  v8f acc[8];
  #pragma unroll
  for (int nc = 0; nc < 8; ++nc) acc[nc] = vz;

  #pragma unroll
  for (int kc = 0; kc < 4; ++kc) {
    const h2* ap = (const h2*)(rb0 + kc * 32);
    v16h a;
    #pragma unroll
    for (int j = 0; j < 8; ++j) {
      h2 t = ap[j + 4 * half + ((j >= 4) ? 4 : 0)];
      a[2*j] = t.x; a[2*j+1] = t.y;
    }
    const v16h* bp = (const v16h*)w0f + ((size_t)kc * 8) * 32 + lane;
    #pragma unroll
    for (int nc = 0; nc < 8; ++nc) {
      v16h b = bp[(size_t)nc * 32];
      acc[nc] = __builtin_amdgcn_wmma_f32_16x16x32_f16(false, a, false, b,
                                                       (short)0, acc[nc], false, false);
    }
  }
  #pragma unroll
  for (int nc = 0; nc < 8; ++nc) {
    int n = nc * 16 + nl;
    float bb = b0[n];
    #pragma unroll
    for (int r = 0; r < 8; ++r) {
      float v = acc[nc][r] + bb; v = v > 0.f ? v : 0.f;
      hp[(r + 8 * half) * 128 + n] = (_Float16)v;
    }
  }
  #pragma unroll
  for (int nc = 0; nc < 8; ++nc) acc[nc] = vz;
  #pragma unroll
  for (int kc = 0; kc < 4; ++kc) {
    v16h a;
    #pragma unroll
    for (int j = 0; j < 8; ++j) {
      int k = kc * 32 + ((j < 4) ? (2*j + 8*half) : (16 + 2*(j-4) + 8*half));
      a[2*j]   = hp[nl * 128 + k];
      a[2*j+1] = hp[nl * 128 + k + 1];
    }
    const v16h* bp = (const v16h*)w1f + ((size_t)kc * 8) * 32 + lane;
    #pragma unroll
    for (int nc = 0; nc < 8; ++nc) {
      v16h b = bp[(size_t)nc * 32];
      acc[nc] = __builtin_amdgcn_wmma_f32_16x16x32_f16(false, a, false, b,
                                                       (short)0, acc[nc], false, false);
    }
  }
  #pragma unroll
  for (int nc = 0; nc < 8; ++nc) {
    int n = nc * 16 + nl;
    float bb = b1[n];
    #pragma unroll
    for (int r = 0; r < 8; ++r) {
      float v = acc[nc][r] + bb; v = v > 0.f ? v : 0.f;
      hp[(r + 8 * half) * 128 + n] = (_Float16)v;
    }
  }
  // last layer: 128 -> 16 (cols 0..2 valid)
  v8f a2 = vz;
  #pragma unroll
  for (int kc = 0; kc < 4; ++kc) {
    v16h a;
    #pragma unroll
    for (int j = 0; j < 8; ++j) {
      int k = kc * 32 + ((j < 4) ? (2*j + 8*half) : (16 + 2*(j-4) + 8*half));
      a[2*j]   = hp[nl * 128 + k];
      a[2*j+1] = hp[nl * 128 + k + 1];
    }
    const v16h* bp = (const v16h*)w2f + ((size_t)kc * 1) * 32 + lane;
    v16h b = *bp;
    a2 = __builtin_amdgcn_wmma_f32_16x16x32_f16(false, a, false, b, (short)0, a2, false, false);
  }
  int n = nl;
  if (n < 3) {
    float sc = ostd[n], mn = omean[n], bb = b2[n];
    #pragma unroll
    for (int r = 0; r < 8; ++r) {
      int grow = tilerow + r + 8 * half;
      out[(size_t)grow * 3 + n] = (a2[r] + bb) * sc + mn;
    }
  }
}

// ---------------------------------------------------------------------------
extern "C" void kernel_launch(void* const* d_in, const int* in_sizes, int n_in,
                              void* d_out, int out_size, void* d_ws, size_t ws_size,
                              hipStream_t stream) {
  (void)in_sizes; (void)n_in; (void)out_size; (void)ws_size;
  auto F = [&](int i){ return (const float*)d_in[i]; };
  auto I = [&](int i){ return (const int*)d_in[i]; };

  // input order: setup_inputs() dict insertion order, nested dicts depth-first
  const float* wp  = F(0);  const float* twp = F(1);  const float* mp = F(2);
  const int* ntp   = I(3);
  const int* srcs  = I(4);  const int* dsts  = I(5);
  const int* wsrcs = I(6);  const int* wdsts = I(7);
  // norms 8..15: node_mean,node_std,mesh_mean,mesh_std,world_mean,world_std,out_mean,out_std
  // enc_node 16..23, enc_mesh 24..31, enc_world 32..39
  // proc.mesh 40..47, proc.world 48..55, proc.node 56..63 (each W0,b0,W1,b1,W2,b2,g,be)
  // dec 64..69

  char* base = (char*)d_ws;
  size_t off = 0;
  auto carve = [&](size_t bytes) -> void* {
    void* p = base + off;
    off += (bytes + 255) & ~(size_t)255;
    return p;
  };

  // weight fragments (f16)
  _Float16* fEnW0 = (_Float16*)carve(4096 * 2);
  _Float16* fEnW1 = (_Float16*)carve(16384 * 2);
  _Float16* fEnW2 = (_Float16*)carve(16384 * 2);
  _Float16* fEmW0 = (_Float16*)carve(4096 * 2);
  _Float16* fEmW1 = (_Float16*)carve(16384 * 2);
  _Float16* fEmW2 = (_Float16*)carve(16384 * 2);
  _Float16* fEwW0 = (_Float16*)carve(4096 * 2);
  _Float16* fEwW1 = (_Float16*)carve(16384 * 2);
  _Float16* fEwW2 = (_Float16*)carve(16384 * 2);
  _Float16* fPMW0 = (_Float16*)carve((size_t)Sn * 49152 * 2);
  _Float16* fPMW1 = (_Float16*)carve((size_t)Sn * 16384 * 2);
  _Float16* fPMW2 = (_Float16*)carve((size_t)Sn * 16384 * 2);
  _Float16* fPWW0 = (_Float16*)carve((size_t)Sn * 49152 * 2);
  _Float16* fPWW1 = (_Float16*)carve((size_t)Sn * 16384 * 2);
  _Float16* fPWW2 = (_Float16*)carve((size_t)Sn * 16384 * 2);
  _Float16* fPNW0 = (_Float16*)carve((size_t)Sn * 49152 * 2);
  _Float16* fPNW1 = (_Float16*)carve((size_t)Sn * 16384 * 2);
  _Float16* fPNW2 = (_Float16*)carve((size_t)Sn * 16384 * 2);
  _Float16* fDW0  = (_Float16*)carve(16384 * 2);
  _Float16* fDW1  = (_Float16*)carve(16384 * 2);
  _Float16* fDW2  = (_Float16*)carve(2048 * 2);
  // features
  _Float16* nfH  = (_Float16*)carve((size_t)Nn * 32 * 2);
  _Float16* mefH = (_Float16*)carve((size_t)EMn * 32 * 2);
  _Float16* wefH = (_Float16*)carve((size_t)EWn * 32 * 2);
  // latent state (f32 residual carrier + f16 GEMM operand)
  float*    node_f = (float*)carve((size_t)Nn * 128 * 4);
  _Float16* node_h = (_Float16*)carve((size_t)Nn * 128 * 2);
  float*    em_f   = (float*)carve((size_t)EMn * 128 * 4);
  _Float16* em_h   = (_Float16*)carve((size_t)EMn * 128 * 2);
  float*    ew_f   = (float*)carve((size_t)EWn * 128 * 4);
  _Float16* ew_h   = (_Float16*)carve((size_t)EWn * 128 * 2);
  // aggregation (agg_m,agg_w contiguous; aggm_h,aggw_h contiguous)
  float*    agg_m  = (float*)carve((size_t)Nn * 128 * 4);
  float*    agg_w  = (float*)carve((size_t)Nn * 128 * 4);
  _Float16* aggm_h = (_Float16*)carve((size_t)Nn * 128 * 2);
  _Float16* aggw_h = (_Float16*)carve((size_t)Nn * 128 * 2);

  auto prep = [&](const float* W, _Float16* frag, int Kact, int Nact, int KC, int NC,
                  int nst, size_t wstride, size_t fstride) {
    size_t total = (size_t)nst * KC * NC * 512;
    prep_w<<<(unsigned)((total + 255) / 256), 256, 0, stream>>>(
        W, frag, Kact, Nact, KC, NC, wstride, fstride, total);
  };

  // ---- weight prep ----
  prep(F(16), fEnW0, 12, 128, 1, 8, 1, 0, 0);
  prep(F(18), fEnW1, 128, 128, 4, 8, 1, 0, 0);
  prep(F(20), fEnW2, 128, 128, 4, 8, 1, 0, 0);
  prep(F(24), fEmW0, 8, 128, 1, 8, 1, 0, 0);
  prep(F(26), fEmW1, 128, 128, 4, 8, 1, 0, 0);
  prep(F(28), fEmW2, 128, 128, 4, 8, 1, 0, 0);
  prep(F(32), fEwW0, 4, 128, 1, 8, 1, 0, 0);
  prep(F(34), fEwW1, 128, 128, 4, 8, 1, 0, 0);
  prep(F(36), fEwW2, 128, 128, 4, 8, 1, 0, 0);
  prep(F(40), fPMW0, 384, 128, 12, 8, Sn, (size_t)384 * 128, 49152);
  prep(F(42), fPMW1, 128, 128, 4, 8, Sn, 16384, 16384);
  prep(F(44), fPMW2, 128, 128, 4, 8, Sn, 16384, 16384);
  prep(F(48), fPWW0, 384, 128, 12, 8, Sn, (size_t)384 * 128, 49152);
  prep(F(50), fPWW1, 128, 128, 4, 8, Sn, 16384, 16384);
  prep(F(52), fPWW2, 128, 128, 4, 8, Sn, 16384, 16384);
  prep(F(56), fPNW0, 384, 128, 12, 8, Sn, (size_t)384 * 128, 49152);
  prep(F(58), fPNW1, 128, 128, 4, 8, Sn, 16384, 16384);
  prep(F(60), fPNW2, 128, 128, 4, 8, Sn, 16384, 16384);
  prep(F(64), fDW0, 128, 128, 4, 8, 1, 0, 0);
  prep(F(66), fDW1, 128, 128, 4, 8, 1, 0, 0);
  prep(F(68), fDW2, 128, 3, 4, 1, 1, 0, 0);

  // ---- features ----
  node_feat<<<Nn / 256, 256, 0, stream>>>(wp, twp, ntp, F(8), F(9), nfH);
  mesh_feat<<<EMn / 256, 256, 0, stream>>>(wp, mp, srcs, dsts, F(10), F(11), mefH);
  world_feat<<<EWn / 256, 256, 0, stream>>>(wp, wsrcs, wdsts, F(12), F(13), wefH);

  // ---- encoders ----
  mlp_fused<1, false, false, false><<<Nn / 128, 256, 0, stream>>>(
      nfH, 32, nullptr, nullptr, nullptr, nullptr,
      fEnW0, fEnW1, fEnW2, F(17), F(19), F(21), F(22), F(23),
      node_f, node_h, nullptr, nullptr);
  mlp_fused<1, false, false, false><<<EMn / 128, 256, 0, stream>>>(
      mefH, 32, nullptr, nullptr, nullptr, nullptr,
      fEmW0, fEmW1, fEmW2, F(25), F(27), F(29), F(30), F(31),
      em_f, em_h, nullptr, nullptr);
  mlp_fused<1, false, false, false><<<EWn / 128, 256, 0, stream>>>(
      wefH, 32, nullptr, nullptr, nullptr, nullptr,
      fEwW0, fEwW1, fEwW2, F(33), F(35), F(37), F(38), F(39),
      ew_f, ew_h, nullptr, nullptr);

  // ---- 15 processor steps ----
  for (int s = 0; s < Sn; ++s) {
    size_t so = (size_t)s * 128;
    zero_f32<<<1024, 256, 0, stream>>>(agg_m, (size_t)2 * Nn * 128);
    // mesh-edge MLP (+residual, +scatter into agg_m[dsts])
    mlp_fused<12, true, true, true><<<EMn / 128, 256, 0, stream>>>(
        em_h, 128, node_h, node_h, srcs, dsts,
        fPMW0 + (size_t)s * 49152, fPMW1 + (size_t)s * 16384, fPMW2 + (size_t)s * 16384,
        F(41) + so, F(43) + so, F(45) + so, F(46) + so, F(47) + so,
        em_f, em_h, agg_m, dsts);
    // world-edge MLP (+residual, +scatter into agg_w[wdsts])
    mlp_fused<12, true, true, true><<<EWn / 128, 256, 0, stream>>>(
        ew_h, 128, node_h, node_h, wsrcs, wdsts,
        fPWW0 + (size_t)s * 49152, fPWW1 + (size_t)s * 16384, fPWW2 + (size_t)s * 16384,
        F(49) + so, F(51) + so, F(53) + so, F(54) + so, F(55) + so,
        ew_f, ew_h, agg_w, wdsts);
    // aggregates -> f16 for node-MLP A operands
    cvt_f32_h<<<1024, 256, 0, stream>>>(agg_m, aggm_h, (size_t)2 * Nn * 128);
    // node MLP (+residual)
    mlp_fused<12, true, true, false><<<Nn / 128, 256, 0, stream>>>(
        node_h, 128, aggm_h, aggw_h, nullptr, nullptr,
        fPNW0 + (size_t)s * 49152, fPNW1 + (size_t)s * 16384, fPNW2 + (size_t)s * 16384,
        F(57) + so, F(59) + so, F(61) + so, F(62) + so, F(63) + so,
        node_f, node_h, nullptr, nullptr);
  }

  // ---- decoder + unnormalization ----
  dec_kernel<<<Nn / 128, 256, 0, stream>>>(
      node_h, fDW0, fDW1, fDW2, F(65), F(67), F(69), F(14), F(15), (float*)d_out);
}
